// DPLSTMCell_35510789603698
// MI455X (gfx1250) — matrix operand order repaired and checked by reading, not compiled, not run on hardware
//
#include <hip/hip_runtime.h>

// CDNA5 / gfx1250 fused LSTM cell:
//   gates = X*W_ih^T + H*W_hh^T + b_ih + b_hh   (bf16 WMMA, fp32 accum)
//   c_t = sig(f)*c_prev + sig(i)*tanh(g); h_t = sig(o)*tanh(c_t)
// d_out = [h_t (B*H) | c_t (B*H)] fp32.
//
// Primary path: pre-pack operands to bf16 in d_ws, then GEMM kernel stages
// tiles with GLOBAL_LOAD_ASYNC_TO_LDS_B128 (ASYNCcnt), double-buffered LDS.
// Fallback path (ws too small): fp32->bf16 convert-on-stage.

typedef __attribute__((ext_vector_type(16))) __bf16 v16bf;
typedef __attribute__((ext_vector_type(8)))  float  v8f;
typedef int v4i_b128 __attribute__((vector_size(16)));

#define B_DIM 8192
#define H_DIM 1024
#define K_DIM 1024        // per-phase K (IN == H == 1024)
#define LDS_STRIDE 72     // 64 bf16 + 8 pad (144 B rows, 16B-aligned)

#if defined(__has_builtin)
#  if __has_builtin(__builtin_amdgcn_global_load_async_to_lds_b128)
#    define HAS_ASYNC_LDS 1
#  endif
#  if __has_builtin(__builtin_amdgcn_s_wait_asynccnt)
#    define HAS_WAIT_ASYNC 1
#  endif
#endif

__device__ __forceinline__ unsigned pack2_bf16_rne(float a, float b) {
  unsigned ua = __float_as_uint(a);
  unsigned ub = __float_as_uint(b);
  ua = (ua + 0x7FFFu + ((ua >> 16) & 1u)) >> 16;
  ub = (ub + 0x7FFFu + ((ub >> 16) & 1u)) >> 16;
  return ua | (ub << 16);
}

__device__ __forceinline__ float fast_sigmoid(float x) {
  return 1.0f / (1.0f + __expf(-x));
}

__device__ __forceinline__ float fast_tanh(float x) {
  float ax = __builtin_fabsf(x);
  float e  = __expf(-2.0f * ax);
  float t  = (1.0f - e) / (1.0f + e);
  return __builtin_copysignf(t, x);
}

union FragBF16 { uint4 u[2]; v16bf v; };

// 16-byte global->LDS stage: async DMA if available, else reg round-trip.
__device__ __forceinline__ void stage16(const unsigned short* g, unsigned short* l) {
#ifdef HAS_ASYNC_LDS
  __builtin_amdgcn_global_load_async_to_lds_b128(
      (__attribute__((address_space(1))) v4i_b128*)g,
      (__attribute__((address_space(3))) v4i_b128*)l, 0, 0);
#else
  *(uint4*)l = *(const uint4*)g;
#endif
}

__device__ __forceinline__ void wait_async_copies() {
#ifdef HAS_ASYNC_LDS
#  ifdef HAS_WAIT_ASYNC
  __builtin_amdgcn_s_wait_asynccnt(0);
#  else
  asm volatile("s_wait_asynccnt 0x0" ::: "memory");
#  endif
#endif
}

// One 64-K slab of MMAs: 2 k-chunks x (2 M-tiles x 4 gate-tiles) = 16 WMMA.
__device__ __forceinline__ void mma_slab(const unsigned short* sA,
                                         const unsigned short* sB,
                                         int mg, int ng, int laneN, int hi,
                                         v8f acc[2][4]) {
#pragma unroll
  for (int kk = 0; kk < 64; kk += 32) {
    FragBF16 fa[2];
#pragma unroll
    for (int mt = 0; mt < 2; ++mt) {
      // A 16x32 bf16: lanes<16 -> K kk+[0..7],[16..23]; lanes>=16 -> +8
      const unsigned short* pa = &sA[(mg * 32 + mt * 16 + laneN) * LDS_STRIDE + kk + hi * 8];
      fa[mt].u[0] = *(const uint4*)(pa);
      fa[mt].u[1] = *(const uint4*)(pa + 16);
    }
#pragma unroll
    for (int g = 0; g < 4; ++g) {
      // B 32x16 bf16: lane n = column n; lanes<16 -> K kk+[0..7],[16..23]; lanes>=16 -> +8
      FragBF16 fb;
      const unsigned short* pb = &sB[(g * 32 + ng * 16 + laneN) * LDS_STRIDE + kk + hi * 8];
      fb.u[0] = *(const uint4*)(pb);
      fb.u[1] = *(const uint4*)(pb + 16);
#pragma unroll
      for (int mt = 0; mt < 2; ++mt) {
        acc[mt][g] = __builtin_amdgcn_wmma_f32_16x16x32_bf16(
            false, fa[mt].v, false, fb.v, (short)0, acc[mt][g], false, false);
      }
    }
  }
}

// Fused LSTM epilogue: each lane owns matching i/f/g/o elements.
__device__ __forceinline__ void lstm_epilogue(v8f acc[2][4],
                                              const float* __restrict__ c_prev,
                                              const float* __restrict__ b_ih,
                                              const float* __restrict__ b_hh,
                                              float* __restrict__ out,
                                              int mBase, int nBase,
                                              int mg, int ng, int laneN, int hi) {
  const int hcol = nBase + ng * 16 + laneN;
  float bias[4];
#pragma unroll
  for (int g = 0; g < 4; ++g)
    bias[g] = b_ih[g * H_DIM + hcol] + b_hh[g * H_DIM + hcol];

  const size_t cOutBase = (size_t)B_DIM * H_DIM;
#pragma unroll
  for (int mt = 0; mt < 2; ++mt) {
    const int rbase = mBase + mg * 32 + mt * 16 + hi * 8;
#pragma unroll
    for (int e = 0; e < 8; ++e) {
      const int row = rbase + e;
      const float iv = fast_sigmoid(acc[mt][0][e] + bias[0]);
      const float fv = fast_sigmoid(acc[mt][1][e] + bias[1]);
      const float gv = fast_tanh   (acc[mt][2][e] + bias[2]);
      const float ov = fast_sigmoid(acc[mt][3][e] + bias[3]);
      const float cp = c_prev[(size_t)row * H_DIM + hcol];
      const float ct = fv * cp + iv * gv;
      const float ht = ov * fast_tanh(ct);
      out[(size_t)row * H_DIM + hcol]            = ht;
      out[cOutBase + (size_t)row * H_DIM + hcol] = ct;
    }
  }
}

// ---------- fp32 -> bf16 pre-pack (primary path helper) ----------
__global__ __launch_bounds__(256)
void pack_bf16_kernel(const float* __restrict__ in, unsigned short* __restrict__ out, int n4) {
  const int i = blockIdx.x * blockDim.x + threadIdx.x;
  if (i < n4) {
    const float4 v = ((const float4*)in)[i];
    ((uint2*)out)[i] = make_uint2(pack2_bf16_rne(v.x, v.y), pack2_bf16_rne(v.z, v.w));
  }
}

// ---------- primary GEMM: bf16 sources, async-LDS double-buffered ----------
__global__ __launch_bounds__(256)
void lstm_cell_wmma_bf16_async(const unsigned short* __restrict__ xc,
                               const unsigned short* __restrict__ hc,
                               const unsigned short* __restrict__ wihc,
                               const unsigned short* __restrict__ whhc,
                               const float* __restrict__ c_prev,
                               const float* __restrict__ b_ih,
                               const float* __restrict__ b_hh,
                               float* __restrict__ out)
{
  __shared__ __align__(16) unsigned short sA[2][128 * LDS_STRIDE];
  __shared__ __align__(16) unsigned short sB[2][128 * LDS_STRIDE];

  const int tid   = threadIdx.x;
  const int wv    = tid >> 5;
  const int lane  = tid & 31;
  const int laneN = lane & 15;
  const int hi    = lane >> 4;
  const int mg    = wv >> 1;
  const int ng    = wv & 1;
  const int mBase = blockIdx.x * 128;
  const int nBase = blockIdx.y * 32;

  // This thread's 4 staging chunks per tile (16 B each): 1024 chunks / 256 thr.
  const int rowc = tid >> 3;          // base row  (0..31), +32 per i
  const int chc  = tid & 7;           // 16B chunk within 64-K slab

  v8f acc[2][4];
#pragma unroll
  for (int mt = 0; mt < 2; ++mt)
#pragma unroll
    for (int g = 0; g < 4; ++g)
#pragma unroll
      for (int e = 0; e < 8; ++e)
        acc[mt][g][e] = 0.0f;

  // Issue all staging copies for flattened step s into buffer b.
  auto issue = [&](int s, int b) {
    const int phase = s >> 4;
    const int ks    = (s & 15) * 64;
    const unsigned short* __restrict__ As = phase ? hc   : xc;
    const unsigned short* __restrict__ Ws = phase ? whhc : wihc;
#pragma unroll
    for (int i = 0; i < 4; ++i) {
      const int row = rowc + i * 32;      // 0..127
      // A tile: activations, row-major [B,1024]
      stage16(As + (size_t)(mBase + row) * K_DIM + ks + chc * 8,
              &sA[b][row * LDS_STRIDE + chc * 8]);
      // B tile: combined-N column `row`: gate = row>>5, h-col = nBase+(row&31)
      const int wrow = ((row >> 5) * H_DIM) + nBase + (row & 31);
      stage16(Ws + (size_t)wrow * K_DIM + ks + chc * 8,
              &sB[b][row * LDS_STRIDE + chc * 8]);
    }
  };

  issue(0, 0);
  wait_async_copies();
  __syncthreads();

#pragma unroll 1
  for (int s = 0; s < 32; ++s) {
    const int buf = s & 1;
    if (s + 1 < 32) issue(s + 1, buf ^ 1);       // prefetch next slab (async)
    mma_slab(sA[buf], sB[buf], mg, ng, laneN, hi, acc);
    wait_async_copies();                          // my async copies landed
    __syncthreads();                              // everyone done read + write
  }

  lstm_epilogue(acc, c_prev, b_ih, b_hh, out, mBase, nBase, mg, ng, laneN, hi);
}

// ---------- fallback GEMM: fp32 sources, convert-on-stage ----------
__global__ __launch_bounds__(256)
void lstm_cell_wmma_f32src(const float* __restrict__ x,
                           const float* __restrict__ h_prev,
                           const float* __restrict__ c_prev,
                           const float* __restrict__ W_ih,
                           const float* __restrict__ b_ih,
                           const float* __restrict__ W_hh,
                           const float* __restrict__ b_hh,
                           float* __restrict__ out)
{
  __shared__ __align__(16) unsigned short sA[128 * LDS_STRIDE];
  __shared__ __align__(16) unsigned short sB[128 * LDS_STRIDE];

  const int tid   = threadIdx.x;
  const int wv    = tid >> 5;
  const int lane  = tid & 31;
  const int laneN = lane & 15;
  const int hi    = lane >> 4;
  const int mg    = wv >> 1;
  const int ng    = wv & 1;
  const int mBase = blockIdx.x * 128;
  const int nBase = blockIdx.y * 32;

  v8f acc[2][4];
#pragma unroll
  for (int mt = 0; mt < 2; ++mt)
#pragma unroll
    for (int g = 0; g < 4; ++g)
#pragma unroll
      for (int e = 0; e < 8; ++e)
        acc[mt][g][e] = 0.0f;

#pragma unroll 1
  for (int phase = 0; phase < 2; ++phase) {
    const float* __restrict__ As = phase ? h_prev : x;
    const float* __restrict__ Ws = phase ? W_hh  : W_ih;
#pragma unroll 1
    for (int ks = 0; ks < K_DIM; ks += 64) {
#pragma unroll
      for (int i = 0; i < 8; ++i) {
        const int id  = tid + i * 256;
        const int row = id >> 4;
        const int kc  = id & 15;
        const float4 a4 = *(const float4*)(As + (size_t)(mBase + row) * K_DIM + ks + kc * 4);
        *(uint2*)(&sA[row * LDS_STRIDE + kc * 4]) =
            make_uint2(pack2_bf16_rne(a4.x, a4.y), pack2_bf16_rne(a4.z, a4.w));
        const int wrow = ((row >> 5) * H_DIM) + nBase + (row & 31);
        const float4 b4 = *(const float4*)(Ws + (size_t)wrow * K_DIM + ks + kc * 4);
        *(uint2*)(&sB[row * LDS_STRIDE + kc * 4]) =
            make_uint2(pack2_bf16_rne(b4.x, b4.y), pack2_bf16_rne(b4.z, b4.w));
      }
      if (ks + 64 < K_DIM) {
        const int prow = tid >> 1;
        const int pk   = ks + 64 + (tid & 1) * 32;
        __builtin_prefetch(As + (size_t)(mBase + prow) * K_DIM + pk, 0, 1);
        __builtin_prefetch(Ws + (size_t)(((prow >> 5) * H_DIM) + nBase + (prow & 31)) * K_DIM + pk, 0, 1);
      }
      __syncthreads();
      mma_slab(sA, sB, mg, ng, laneN, hi, acc);
      __syncthreads();
    }
  }

  lstm_epilogue(acc, c_prev, b_ih, b_hh, out, mBase, nBase, mg, ng, laneN, hi);
}

extern "C" void kernel_launch(void* const* d_in, const int* in_sizes, int n_in,
                              void* d_out, int out_size, void* d_ws, size_t ws_size,
                              hipStream_t stream) {
  (void)in_sizes; (void)n_in; (void)out_size;
  const float* x      = (const float*)d_in[0];
  const float* h_prev = (const float*)d_in[1];
  const float* c_prev = (const float*)d_in[2];
  const float* W_ih   = (const float*)d_in[3];
  const float* b_ih   = (const float*)d_in[4];
  const float* W_hh   = (const float*)d_in[5];
  const float* b_hh   = (const float*)d_in[6];
  float* out = (float*)d_out;

  const dim3 grid(B_DIM / 128, H_DIM / 32);   // 64 x 32 workgroups
  const dim3 block(256);                      // 8 waves (wave32)

  const size_t actElems = (size_t)B_DIM * K_DIM;          // 8M
  const size_t wElems   = (size_t)4 * H_DIM * K_DIM;      // 4M
  const size_t needWs   = (2 * actElems + 2 * wElems) * sizeof(unsigned short); // 48 MB

  if (ws_size >= needWs) {
    unsigned short* xc   = (unsigned short*)d_ws;
    unsigned short* hc   = xc + actElems;
    unsigned short* wihc = hc + actElems;
    unsigned short* whhc = wihc + wElems;

    const int n4a = (int)(actElems / 4);
    const int n4w = (int)(wElems / 4);
    hipLaunchKernelGGL(pack_bf16_kernel, dim3((n4a + 255) / 256), dim3(256), 0, stream, x,      xc,   n4a);
    hipLaunchKernelGGL(pack_bf16_kernel, dim3((n4a + 255) / 256), dim3(256), 0, stream, h_prev, hc,   n4a);
    hipLaunchKernelGGL(pack_bf16_kernel, dim3((n4w + 255) / 256), dim3(256), 0, stream, W_ih,   wihc, n4w);
    hipLaunchKernelGGL(pack_bf16_kernel, dim3((n4w + 255) / 256), dim3(256), 0, stream, W_hh,   whhc, n4w);

    hipLaunchKernelGGL(lstm_cell_wmma_bf16_async, grid, block, 0, stream,
                       xc, hc, wihc, whhc, c_prev, b_ih, b_hh, out);
  } else {
    hipLaunchKernelGGL(lstm_cell_wmma_f32src, grid, block, 0, stream,
                       x, h_prev, c_prev, W_ih, b_ih, W_hh, b_hh, out);
  }
}
